// CausalSelfAttention_59365037965942
// MI455X (gfx1250) — matrix operand order repaired and checked by reading, not compiled, not run on hardware
//
#include <hip/hip_runtime.h>
#include <hip/hip_bf16.h>

// ---------------------------------------------------------------------------
// Causal self-attention forward, bf16 WMMA pipeline for gfx1250 (MI455X).
//   B=2, S=4096, D=512, H=8, DK=64
// cvt(f32->bf16) -> QKV GEMMs (WMMA) -> flash attention (WMMA + online
// softmax + async-to-LDS staging) -> output GEMM (WMMA) -> f32 out.
// ---------------------------------------------------------------------------

constexpr int Bb  = 2;
constexpr int Ss  = 4096;
constexpr int Dd  = 512;
constexpr int Hh  = 8;
constexpr int DKk = 64;

typedef __attribute__((ext_vector_type(16))) __bf16 v16bf;
typedef __attribute__((ext_vector_type(8)))  __bf16 v8bf;
typedef __attribute__((ext_vector_type(8)))  float  v8f;
typedef __attribute__((ext_vector_type(4)))  int    v4i;

// Pointers in explicit AMDGPU target address spaces for the async-DMA builtin
// (param 0 is AS1 "global", param 1 is AS3 "LDS" per the clang diagnostic).
typedef v4i __attribute__((address_space(1)))* gptr128;
typedef v4i __attribute__((address_space(3)))* lptr128;

#if defined(__has_builtin)
#if __has_builtin(__builtin_amdgcn_global_load_async_to_lds_b128)
#define HAVE_ASYNC_LDS 1
#endif
#endif
#ifndef HAVE_ASYNC_LDS
#define HAVE_ASYNC_LDS 0
#endif

__device__ __forceinline__ __bf16 f2bf(float f) {
  union { float f; unsigned u; } v; v.f = f;
  unsigned r = (v.u + 0x7FFFu + ((v.u >> 16) & 1u)) >> 16;   // RNE
  unsigned short s = (unsigned short)r;
  __bf16 o; __builtin_memcpy(&o, &s, 2);
  return o;
}

__device__ __forceinline__ v16bf cat8(v8bf lo, v8bf hi) {
  return __builtin_shufflevector(lo, hi, 0,1,2,3,4,5,6,7,8,9,10,11,12,13,14,15);
}

// WMMA A fragment (16 rows x 32 K, bf16), source row-major [m][k], stride ldk.
__device__ __forceinline__ v16bf load_a_frag(const __bf16* rowbase, int ldk,
                                             int k0, int lane) {
  const __bf16* p = rowbase + (lane & 15) * ldk + k0 + ((lane >> 4) << 3);
  v8bf lo = *(const v8bf*)p;
  v8bf hi = *(const v8bf*)(p + 16);
  return cat8(lo, hi);
}

// WMMA B fragment (32 K x 16 N, bf16) from *transposed* storage [n][k].
__device__ __forceinline__ v16bf load_b_frag(const __bf16* nbase, int ldk,
                                             int k0, int lane) {
  const __bf16* p = nbase + (lane & 15) * ldk + k0 + ((lane >> 4) << 3);
  v8bf lo = *(const v8bf*)p;
  v8bf hi = *(const v8bf*)(p + 16);
  return cat8(lo, hi);
}

__device__ __forceinline__ v8f wmma_bf16(v16bf a, v16bf b, v8f c) {
  return __builtin_amdgcn_wmma_f32_16x16x32_bf16(false, a, false, b,
                                                 (short)0, c, false, false);
}

// 16-byte global -> LDS copy: async DMA (ASYNCcnt) if available, else direct.
__device__ __forceinline__ void cp_b128_to_lds(__bf16* lds, const __bf16* g) {
#if HAVE_ASYNC_LDS
  __builtin_amdgcn_global_load_async_to_lds_b128(
      (gptr128)(v4i*)(void*)const_cast<__bf16*>(g),
      (lptr128)(v4i*)(void*)lds, 0, 0);
#else
  *(v8bf*)lds = *(const v8bf*)g;
#endif
}

__device__ __forceinline__ void wait_async0() {
#if HAVE_ASYNC_LDS
#if __has_builtin(__builtin_amdgcn_s_wait_asynccnt)
  __builtin_amdgcn_s_wait_asynccnt(0);
#else
  asm volatile("s_wait_asynccnt 0x0" ::: "memory");
#endif
#endif
}

// ---------------------------------------------------------------------------
// f32 -> bf16 conversion
// ---------------------------------------------------------------------------
__global__ void cvt_f32_bf16(const float* __restrict__ in,
                             __bf16* __restrict__ out, int n) {
  int i = blockIdx.x * blockDim.x + threadIdx.x;
  int stride = gridDim.x * blockDim.x;
  for (; i < n; i += stride) out[i] = f2bf(in[i]);
}

// ---------------------------------------------------------------------------
// Y = A(8192x512) @ W^T(512x512) + bias.  W stored (N,K) row-major == the
// transposed-B layout the WMMA B fragment wants; both operands stream from
// global (W is L2-resident).  Each wave: 32 rows x 64 cols (8 C tiles), so
// every B fragment feeds two independent WMMAs -> load latency hidden.
// headMode=1: bf16 out in (b,h,s,dk).  headMode=0: f32 out (B,S,D).
// ---------------------------------------------------------------------------
__global__ __launch_bounds__(128) void gemm512(
    const __bf16* __restrict__ A, const __bf16* __restrict__ Wt,
    const float* __restrict__ bias, __bf16* __restrict__ outHead,
    float* __restrict__ outF, int headMode) {
  const int lane = threadIdx.x & 31;
  const int wave = threadIdx.x >> 5;
  const int half = lane >> 4;
  const int m0 = blockIdx.y * 128 + wave * 32;
  const int n0 = blockIdx.x * 64;

  v8f acc[2][4] = {};
  const __bf16* arow0 = A + (size_t)m0 * Dd;
  const __bf16* arow1 = A + (size_t)(m0 + 16) * Dd;

  for (int kk = 0; kk < Dd; kk += 32) {
    if (kk + 64 < Dd) {
      __builtin_prefetch(arow0 + (lane & 15) * Dd + kk + 64, 0, 0);
      __builtin_prefetch(arow1 + (lane & 15) * Dd + kk + 64, 0, 0);
    }
    v16bf a0 = load_a_frag(arow0, Dd, kk, lane);
    v16bf a1 = load_a_frag(arow1, Dd, kk, lane);
#pragma unroll
    for (int t = 0; t < 4; ++t) {
      v16bf b = load_b_frag(Wt + (size_t)(n0 + t * 16) * Dd, Dd, kk, lane);
      acc[0][t] = wmma_bf16(a0, b, acc[0][t]);
      acc[1][t] = wmma_bf16(a1, b, acc[1][t]);
    }
  }

#pragma unroll
  for (int r = 0; r < 2; ++r) {
#pragma unroll
    for (int t = 0; t < 4; ++t) {
      const int n = n0 + t * 16 + (lane & 15);
      const float bv = bias[n];
#pragma unroll
      for (int i = 0; i < 8; ++i) {
        const int m = m0 + r * 16 + i + half * 8;  // C: VGPR i -> row i+8*half
        const float v = acc[r][t][i] + bv;
        if (headMode) {
          const int bb = m >> 12, ss = m & (Ss - 1);
          const int hh = n >> 6, dk = n & 63;
          outHead[(((size_t)(bb * Hh + hh)) * Ss + ss) * DKk + dk] = f2bf(v);
        } else {
          outF[(size_t)m * Dd + n] = v;
        }
      }
    }
  }
}

// ---------------------------------------------------------------------------
// Flash attention: one block = 4 waves = 64 query rows of one (b,h).
// K tile staged to LDS via async DMA (natural (key,dk) == transposed-B layout
// for Q*K^T); V staged transposed (dk,key) for P*V; P bounced through
// per-wave LDS to convert C layout -> A layout.  Online softmax row stats via
// 16-lane shfl reductions matching the WMMA C layout.
// ---------------------------------------------------------------------------
__global__ __launch_bounds__(128) void flash64(
    const __bf16* __restrict__ Qg, const __bf16* __restrict__ Kg,
    const __bf16* __restrict__ Vg, __bf16* __restrict__ Og) {
  __shared__ __bf16 Ks[64 * 64];      // (key, dk)
  __shared__ __bf16 Vt[64 * 64];      // (dk, key)
  __shared__ __bf16 Ps[4][16 * 64];   // per-wave P (m, key)

  const int lane = threadIdx.x & 31;
  const int wave = threadIdx.x >> 5;
  const int half = lane >> 4;
  const int qt = blockIdx.x, h = blockIdx.y, b = blockIdx.z;
  const int q_base = qt * 64;

  const size_t headoff = ((size_t)(b * Hh + h)) * Ss * DKk;
  const __bf16* Qh = Qg + headoff;
  const __bf16* Kh = Kg + headoff;
  const __bf16* Vh = Vg + headoff;

  const __bf16* qrow = Qh + (size_t)(q_base + wave * 16) * DKk;
  const v16bf aQ0 = load_a_frag(qrow, DKk, 0, lane);
  const v16bf aQ1 = load_a_frag(qrow, DKk, 32, lane);

  float mrow[8], lrow[8];
#pragma unroll
  for (int i = 0; i < 8; ++i) { mrow[i] = -__builtin_inff(); lrow[i] = 0.f; }
  v8f accO[4] = {};
  const float Cst = 0.18033688011112042f;  // (1/sqrt(DK)) * log2(e)

  for (int kt = 0; kt <= qt; ++kt) {
    const int k_base = kt * 64;
    __syncthreads();  // previous iteration done with Ks/Vt
    {
      const int row = threadIdx.x >> 1;        // 0..63
      const int cb = (threadIdx.x & 1) * 32;   // 0 / 32
      const __bf16* gK = Kh + (size_t)(k_base + row) * DKk + cb;
      __bf16* lK = &Ks[row * 64 + cb];
#pragma unroll
      for (int c = 0; c < 4; ++c)              // 4 x b128 async per thread
        cp_b128_to_lds(lK + c * 8, gK + c * 8);
      const __bf16* gV = Vh + (size_t)(k_base + row) * DKk + cb;
#pragma unroll
      for (int c = 0; c < 4; ++c) {            // V transposed via VGPR bounce
        v8bf vv = *(const v8bf*)(gV + c * 8);
#pragma unroll
        for (int j = 0; j < 8; ++j) Vt[(cb + c * 8 + j) * 64 + row] = vv[j];
      }
    }
    wait_async0();     // ASYNCcnt==0 for this wave's DMA before the barrier
    __syncthreads();

    // S = Q K^T  (16 x 64 per wave, f32 accum)
    v8f sc[4] = {};
#pragma unroll
    for (int t = 0; t < 4; ++t) {
      v16bf b0 = load_b_frag(&Ks[(t * 16) * 64], 64, 0, lane);
      sc[t] = wmma_bf16(aQ0, b0, sc[t]);
      v16bf b1 = load_b_frag(&Ks[(t * 16) * 64], 64, 32, lane);
      sc[t] = wmma_bf16(aQ1, b1, sc[t]);
    }

    if (kt == qt) {  // causal mask, diagonal tile only
#pragma unroll
      for (int t = 0; t < 4; ++t) {
        const int kx = k_base + t * 16 + (lane & 15);
#pragma unroll
        for (int i = 0; i < 8; ++i) {
          const int qx = q_base + wave * 16 + i + half * 8;
          if (kx > qx) sc[t][i] = -__builtin_inff();
        }
      }
    }

    // Online softmax (row = (i, half); reduce across the 16-lane half).
    float mnew[8], alpha[8];
#pragma unroll
    for (int i = 0; i < 8; ++i) {
      float mx = fmaxf(fmaxf(sc[0][i], sc[1][i]), fmaxf(sc[2][i], sc[3][i]));
#pragma unroll
      for (int off = 1; off < 16; off <<= 1)
        mx = fmaxf(mx, __shfl_xor(mx, off, 16));
      mnew[i] = fmaxf(mrow[i], mx);
      alpha[i] = exp2f((mrow[i] - mnew[i]) * Cst);
      mrow[i] = mnew[i];
    }
#pragma unroll
    for (int t = 0; t < 4; ++t)
#pragma unroll
      for (int i = 0; i < 8; ++i)
        sc[t][i] = exp2f((sc[t][i] - mnew[i]) * Cst);
#pragma unroll
    for (int i = 0; i < 8; ++i) {
      float rs = sc[0][i] + sc[1][i] + sc[2][i] + sc[3][i];
#pragma unroll
      for (int off = 1; off < 16; off <<= 1)
        rs += __shfl_xor(rs, off, 16);
      lrow[i] = lrow[i] * alpha[i] + rs;
    }
#pragma unroll
    for (int t = 0; t < 4; ++t)
#pragma unroll
      for (int i = 0; i < 8; ++i) accO[t][i] *= alpha[i];

    // P: C layout -> row-major LDS (wave-private) -> A fragments.
#pragma unroll
    for (int t = 0; t < 4; ++t)
#pragma unroll
      for (int i = 0; i < 8; ++i)
        Ps[wave][(i + half * 8) * 64 + t * 16 + (lane & 15)] = f2bf(sc[t][i]);
    __syncthreads();

    const v16bf aP0 = load_a_frag(&Ps[wave][0], 64, 0, lane);
    const v16bf aP1 = load_a_frag(&Ps[wave][0], 64, 32, lane);
#pragma unroll
    for (int t = 0; t < 4; ++t) {
      v16bf bv0 = load_b_frag(&Vt[(t * 16) * 64], 64, 0, lane);
      accO[t] = wmma_bf16(aP0, bv0, accO[t]);
      v16bf bv1 = load_b_frag(&Vt[(t * 16) * 64], 64, 32, lane);
      accO[t] = wmma_bf16(aP1, bv1, accO[t]);
    }
  }

  // Normalize; write O in (B,S,D) bf16 for the output projection.
#pragma unroll
  for (int i = 0; i < 8; ++i) lrow[i] = 1.0f / lrow[i];
#pragma unroll
  for (int t = 0; t < 4; ++t) {
    const int d = h * DKk + t * 16 + (lane & 15);
#pragma unroll
    for (int i = 0; i < 8; ++i) {
      const int s = q_base + wave * 16 + i + half * 8;
      Og[((size_t)b * Ss + s) * Dd + d] = f2bf(accO[t][i] * lrow[i]);
    }
  }
}

// ---------------------------------------------------------------------------
// Launch
// ---------------------------------------------------------------------------
extern "C" void kernel_launch(void* const* d_in, const int* in_sizes, int n_in,
                              void* d_out, int out_size, void* d_ws,
                              size_t ws_size, hipStream_t stream) {
  const float* x  = (const float*)d_in[0];
  const float* Wq = (const float*)d_in[1];
  const float* bq = (const float*)d_in[2];
  const float* Wk = (const float*)d_in[3];
  const float* bk = (const float*)d_in[4];
  const float* Wv = (const float*)d_in[5];
  const float* bv = (const float*)d_in[6];
  const float* Wo = (const float*)d_in[7];
  const float* bo = (const float*)d_in[8];

  const size_t NX = (size_t)Bb * Ss * Dd;  // 4,194,304
  const size_t NW = (size_t)Dd * Dd;       // 262,144

  __bf16* xb  = (__bf16*)d_ws;
  __bf16* Wqb = xb + NX;
  __bf16* Wkb = Wqb + NW;
  __bf16* Wvb = Wkb + NW;
  __bf16* Wob = Wvb + NW;
  __bf16* Qw  = Wob + NW;   // (b,h,s,dk)
  __bf16* Kw  = Qw + NX;
  __bf16* Vw  = Kw + NX;
  __bf16* Ow  = Vw + NX;    // (b,s,d)

  cvt_f32_bf16<<<2048, 256, 0, stream>>>(x, xb, (int)NX);
  cvt_f32_bf16<<<256, 256, 0, stream>>>(Wq, Wqb, (int)NW);
  cvt_f32_bf16<<<256, 256, 0, stream>>>(Wk, Wkb, (int)NW);
  cvt_f32_bf16<<<256, 256, 0, stream>>>(Wv, Wvb, (int)NW);
  cvt_f32_bf16<<<256, 256, 0, stream>>>(Wo, Wob, (int)NW);

  dim3 gproj(Dd / 64, (Bb * Ss) / 128);  // (8, 64)
  gemm512<<<gproj, 128, 0, stream>>>(xb, Wqb, bq, Qw, nullptr, 1);
  gemm512<<<gproj, 128, 0, stream>>>(xb, Wkb, bk, Kw, nullptr, 1);
  gemm512<<<gproj, 128, 0, stream>>>(xb, Wvb, bv, Vw, nullptr, 1);

  flash64<<<dim3(Ss / 64, Hh, Bb), 128, 0, stream>>>(Qw, Kw, Vw, Ow);

  gemm512<<<gproj, 128, 0, stream>>>(Ow, Wob, bo, nullptr, (float*)d_out, 0);
}
